// WaveNet_1425929142642
// MI455X (gfx1250) — matrix operand order repaired and checked by reading, not compiled, not run on hardware
//
#include <hip/hip_runtime.h>
#include <hip/hip_bf16.h>

typedef __attribute__((ext_vector_type(16))) __bf16 v16bf;
typedef __attribute__((ext_vector_type(8)))  __bf16 v8bf;
typedef __attribute__((ext_vector_type(8)))  float  v8f;

// ---------------- problem constants ----------------
constexpr int T_    = 16384;
constexpr int BATCH = 4;
constexpr int RC    = 32;    // residual channels
constexpr int DC    = 32;    // dilation channels
constexpr int COND  = 80;    // conditioning channels
constexpr int SC    = 256;   // skip channels
constexpr int CLS   = 256;   // classes
constexpr int LAY   = 40;    // layers

// ---------------- tiling ----------------
constexpr int HALO  = 48;            // >= LAY*(K-1)=40, multiple of 16
constexpr int VALID = 128;           // valid columns per tile (divides T)
constexpr int COLS  = HALO + VALID;  // 176 total columns per tile
constexpr int NBLK  = COLS / 16;     // 11 column blocks of 16
constexpr int VBLK0 = HALO / 16;     // first valid block = 3
constexpr int NW    = 4;             // waves per workgroup
constexpr int NTHR  = NW * 32;       // 128 threads
constexpr int MAXB  = 3;             // max column blocks per wave
constexpr int NT    = T_ / VALID;    // 128 tiles per batch

// column-major LDS strides (elements). Per-column stride is a multiple of 8
// elements (16B -> b128 aligned) and the dword stride hits 16 distinct banks.
constexpr int LDK_R = 40;    // s_res: K=32 (+guard pad); 80B/col
constexpr int LDK_A = 40;    // s_act: K=32
constexpr int LDK_H = 104;   // s_h:   K=96 (80 real + 16 zero pad); 208B/col
constexpr int LDK_E = 264;   // end-stage e scratch: K=256; 528B/col

// ---------------- packed A-operand tile indexing ----------------
// per-layer: [dil t0m0,t0m1,t1m0,t1m1][f m0,m1][g m0,m1][cf ks0..2 x m0,m1]
//            [cg ks0..2 x m0,m1][out m0,m1] = 22 tiles
constexpr int TPL     = 22;
constexpr int TB_END1 = LAY * TPL;          // 880
constexpr int TB_END2 = TB_END1 + 16;       // 896
constexpr int NTILES  = TB_END2 + 128;      // 1024
constexpr size_t WS_NEEDED = (size_t)NTILES * 512 * sizeof(__bf16);  // 1 MB

// ---------------- math helpers ----------------
__device__ __forceinline__ float fast_tanh(float x) {
#if __has_builtin(__builtin_amdgcn_tanhf)
  return __builtin_amdgcn_tanhf(x);          // v_tanh_f32
#elif __has_builtin(__builtin_amdgcn_tanh_f32)
  return __builtin_amdgcn_tanh_f32(x);
#else
  const float e2 = __expf(2.f * x);
  return (e2 - 1.f) / (e2 + 1.f);
#endif
}
__device__ __forceinline__ float fast_sigmoid(float x) {
  return 0.5f + 0.5f * fast_tanh(0.5f * x);
}

// ---------------- WMMA helpers ----------------
__device__ __forceinline__ v8f wmma_bf(v16bf a, v16bf b, v8f c) {
  return __builtin_amdgcn_wmma_f32_16x16x32_bf16(
      false, a, false, b, (short)0, c, false, false);
}

__device__ __forceinline__ v16bf join16(v8bf lo, v8bf hi) {
  return __builtin_shufflevector(lo, hi, 0, 1, 2, 3, 4, 5, 6, 7,
                                 8, 9, 10, 11, 12, 13, 14, 15);
}

// A operand gathered from global f32 weights (pack kernel + fallback path).
__device__ __forceinline__ v16bf load_A_gather(const float* __restrict__ W,
                                               int rs, int cs, int m0, int k0,
                                               int kmax, int lane) {
  const int g = (lane >> 4) & 1;
  const int m = m0 + (lane & 15);
  v16bf a;
#pragma unroll
  for (int i = 0; i < 16; ++i) {
    const int k = k0 + ((i & 8) ? 16 : 0) + g * 8 + (i & 7);
    const float v = (k < kmax) ? W[m * rs + k * cs] : 0.f;
    a[i] = (__bf16)v;
  }
  return a;
}

// A operand from pre-packed blob: lane's 16 bf16 are 32 contiguous bytes.
__device__ __forceinline__ v16bf load_A_packed(const __bf16* __restrict__ wp,
                                               int tile, int lane) {
  const __bf16* p = wp + (size_t)tile * 512 + lane * 16;
  const v8bf lo = *(const v8bf*)p;
  const v8bf hi = *(const v8bf*)(p + 8);
  return join16(lo, hi);
}

// B operand from column-major LDS [col][ldK] -> 2 x ds_load_b128.
__device__ __forceinline__ v16bf load_B(const __bf16* buf, int ldK,
                                        int k0, int c0, int lane) {
  const int g = (lane >> 4) & 1;
  const int n = lane & 15;
  const __bf16* p = buf + (c0 + n) * ldK + k0 + g * 8;
  const v8bf lo = *(const v8bf*)p;
  const v8bf hi = *(const v8bf*)(p + 16);
  return join16(lo, hi);
}

// C/D tile (f32) -> bf16 column-major LDS -> 4 cvt_pk + 1 ds_store_b128.
__device__ __forceinline__ void store_C(__bf16* buf, int ldK, int m0, int c0,
                                        const v8f c, int lane) {
  const int g = (lane >> 4) & 1;
  const int n = lane & 15;
  v8bf pk;
#pragma unroll
  for (int v = 0; v < 8; ++v) pk[v] = (__bf16)c[v];
  *(v8bf*)(buf + (c0 + n) * ldK + m0 + g * 8) = pk;
}

// ---------------- weight pack kernel: one wave per A tile ----------------
__global__ __launch_bounds__(32)
void pack_weights_kernel(const float* __restrict__ dil_w,
                         const float* __restrict__ f_w,
                         const float* __restrict__ g_w,
                         const float* __restrict__ cf_w,
                         const float* __restrict__ cg_w,
                         const float* __restrict__ out_w,
                         const float* __restrict__ end1_w,
                         const float* __restrict__ end2_w,
                         __bf16* __restrict__ wp) {
  const int tile = blockIdx.x;
  const int lane = threadIdx.x;
  v16bf a;
  if (tile < TB_END1) {
    const int l = tile / TPL, r = tile % TPL;
    if (r < 4) {
      const int tap = r >> 1, mt = r & 1;
      a = load_A_gather(dil_w + (size_t)l * DC * RC * 2 + tap, RC * 2, 2,
                        mt * 16, 0, 32, lane);
    } else if (r < 6) {
      a = load_A_gather(f_w + (size_t)l * DC * DC, DC, 1, (r - 4) * 16, 0, 32, lane);
    } else if (r < 8) {
      a = load_A_gather(g_w + (size_t)l * DC * DC, DC, 1, (r - 6) * 16, 0, 32, lane);
    } else if (r < 14) {
      const int q = r - 8, ks = q >> 1, mt = q & 1;
      a = load_A_gather(cf_w + (size_t)l * DC * COND, COND, 1, mt * 16,
                        ks * 32, COND, lane);
    } else if (r < 20) {
      const int q = r - 14, ks = q >> 1, mt = q & 1;
      a = load_A_gather(cg_w + (size_t)l * DC * COND, COND, 1, mt * 16,
                        ks * 32, COND, lane);
    } else {
      a = load_A_gather(out_w + (size_t)l * RC * DC, DC, 1, (r - 20) * 16, 0, 32, lane);
    }
  } else if (tile < TB_END2) {
    a = load_A_gather(end1_w, RC, 1, (tile - TB_END1) * 16, 0, 32, lane);
  } else {
    const int q = tile - TB_END2, mt = q >> 3, ks = q & 7;
    a = load_A_gather(end2_w, SC, 1, mt * 16, ks * 32, 256, lane);
  }
  __bf16* p = wp + (size_t)tile * 512 + lane * 16;
  *(v8bf*)p       = __builtin_shufflevector(a, a, 0, 1, 2, 3, 4, 5, 6, 7);
  *(v8bf*)(p + 8) = __builtin_shufflevector(a, a, 8, 9, 10, 11, 12, 13, 14, 15);
}

// ---------------- main kernel ----------------
template <bool PACKED>
__global__ __launch_bounds__(NTHR)
void wavenet_wmma_kernel(const float* __restrict__ x,
                         const float* __restrict__ h,
                         const float* __restrict__ start_w,
                         const float* __restrict__ dil_w,
                         const float* __restrict__ f_w,
                         const float* __restrict__ g_w,
                         const float* __restrict__ cf_w,
                         const float* __restrict__ cg_w,
                         const float* __restrict__ out_w,
                         const float* __restrict__ end1_w,
                         const float* __restrict__ end2_w,
                         const __bf16* __restrict__ wp,
                         float* __restrict__ out) {
  // column-major (K contiguous per column) activation buffers
  __shared__ alignas(16) __bf16 s_h[COLS * LDK_H];         // 36608 B
  __shared__ alignas(16) __bf16 s_res[(COLS + 1) * LDK_R]; // 14160 B (guard col 0)
  __shared__ alignas(16) __bf16 s_act[COLS * LDK_A];       // 14080 B

  const int tid  = threadIdx.x;
  const int lane = tid & 31;
  // scalarized wave id -> per-block guards become scalar branches and EXEC
  // provably stays all-ones at every WMMA.
  const int wave = __builtin_amdgcn_readfirstlane(tid >> 5);
  const int tile = blockIdx.x % NT;
  const int bidx = blockIdx.x / NT;
  const int tbase = tile * VALID - HALO;   // global t of local column 0

  const int g = (lane >> 4) & 1;
  const int n = lane & 15;

  const v8f vzero = {0.f, 0.f, 0.f, 0.f, 0.f, 0.f, 0.f, 0.f};

  // ---- stage h tile (bf16, column-major): coalesced reads, no div/mod ----
  for (int r = wave; r < COND; r += NW) {
    const size_t rowbase = ((size_t)(bidx * COND + r)) * T_;
    for (int c = lane; c < COLS; c += 32) {
      const int t = tbase + c;
      const float v = (t >= 0 && t < T_) ? h[rowbase + t] : 0.f;
      s_h[c * LDK_H + r] = (__bf16)v;
    }
  }
  // zero rows 80..95 of every column: two aligned 16B stores per column
  {
    const v8bf bz = {};
    for (int c = tid; c < COLS; c += NTHR) {
      *(v8bf*)(s_h + c * LDK_H + COND)     = bz;
      *(v8bf*)(s_h + c * LDK_H + COND + 8) = bz;
    }
    if (tid < LDK_R / 8)                   // causal zero guard column (slot 0)
      *(v8bf*)(s_res + tid * 8) = bz;
  }

  // ---- per-wave column-block ownership ----
  int blkid[MAXB];
#pragma unroll
  for (int j = 0; j < MAXB; ++j) blkid[j] = wave + j * NW;

  // ---- start 1x1 conv: residual = start_w * x ; skips = 0 ----
  v8f resC[MAXB][2], skipC[MAXB][2];
#pragma unroll
  for (int j = 0; j < MAXB; ++j) {
    if (blkid[j] >= NBLK) continue;
    const int cb = blkid[j] * 16;
    const int t = tbase + cb + n;
    const float xv = (t >= 0 && t < T_) ? x[(size_t)bidx * T_ + t] : 0.f;
#pragma unroll
    for (int mt = 0; mt < 2; ++mt) {
      v8f r0;
#pragma unroll
      for (int e = 0; e < 8; ++e) {
        const int m = mt * 16 + g * 8 + e;
        r0[e] = start_w[m] * xv;
        skipC[j][mt][e] = 0.f;
      }
      resC[j][mt] = r0;
      store_C(s_res, LDK_R, mt * 16, cb + 1, r0, lane);
    }
  }
  __syncthreads();

  // ---- 40-layer stack ----
#pragma unroll 1
  for (int l = 0; l < LAY; ++l) {
    const int tb = l * TPL;
    const float* dw  = dil_w + (size_t)l * DC * RC * 2;
    const float* fw  = f_w   + (size_t)l * DC * DC;
    const float* gw  = g_w   + (size_t)l * DC * DC;
    const float* cfw = cf_w  + (size_t)l * DC * COND;
    const float* cgw = cg_w  + (size_t)l * DC * COND;
    const float* ow  = out_w + (size_t)l * RC * DC;

    // Phase 1: y = W_tap0 @ res(t-1) + W_tap1 @ res(t)  -> s_act
    {
      v16bf a00, a01, a10, a11;
      if constexpr (PACKED) {
        a00 = load_A_packed(wp, tb + 0, lane);
        a01 = load_A_packed(wp, tb + 1, lane);
        a10 = load_A_packed(wp, tb + 2, lane);
        a11 = load_A_packed(wp, tb + 3, lane);
      } else {
        a00 = load_A_gather(dw + 0, RC * 2, 2, 0, 0, 32, lane);
        a01 = load_A_gather(dw + 0, RC * 2, 2, 16, 0, 32, lane);
        a10 = load_A_gather(dw + 1, RC * 2, 2, 0, 0, 32, lane);
        a11 = load_A_gather(dw + 1, RC * 2, 2, 16, 0, 32, lane);
      }
#pragma unroll
      for (int j = 0; j < MAXB; ++j) {
        if (blkid[j] >= NBLK) continue;
        const int cb = blkid[j] * 16;
        const v16bf bm = load_B(s_res, LDK_R, 0, cb, lane);      // shift -1
        const v16bf b0 = load_B(s_res, LDK_R, 0, cb + 1, lane);  // shift  0
        v8f y0 = wmma_bf(a00, bm, vzero);
        y0 = wmma_bf(a10, b0, y0);
        v8f y1 = wmma_bf(a01, bm, vzero);
        y1 = wmma_bf(a11, b0, y1);
        store_C(s_act, LDK_A, 0, cb, y0, lane);
        store_C(s_act, LDK_A, 16, cb, y1, lane);
      }
    }
    __syncthreads();

    // Phase 2: F = f_w@y + cf_w@h ; G = g_w@y + cg_w@h
    // Fully unrolled K-steps (y, h0, h1, h2): first step folds C into inline 0.
    v8f accF[MAXB][2], accG[MAXB][2];
#pragma unroll
    for (int ks = 0; ks < 4; ++ks) {
      v16bf af0, af1, ag0, ag1;
      const __bf16* bbuf;
      int bk0, bld;
      if (ks == 0) {
        if constexpr (PACKED) {
          af0 = load_A_packed(wp, tb + 4, lane);
          af1 = load_A_packed(wp, tb + 5, lane);
          ag0 = load_A_packed(wp, tb + 6, lane);
          ag1 = load_A_packed(wp, tb + 7, lane);
        } else {
          af0 = load_A_gather(fw, DC, 1, 0, 0, 32, lane);
          af1 = load_A_gather(fw, DC, 1, 16, 0, 32, lane);
          ag0 = load_A_gather(gw, DC, 1, 0, 0, 32, lane);
          ag1 = load_A_gather(gw, DC, 1, 16, 0, 32, lane);
        }
        bbuf = s_act; bk0 = 0; bld = LDK_A;
      } else {
        const int kc = ks - 1;
        if constexpr (PACKED) {
          af0 = load_A_packed(wp, tb + 8 + kc * 2 + 0, lane);
          af1 = load_A_packed(wp, tb + 8 + kc * 2 + 1, lane);
          ag0 = load_A_packed(wp, tb + 14 + kc * 2 + 0, lane);
          ag1 = load_A_packed(wp, tb + 14 + kc * 2 + 1, lane);
        } else {
          af0 = load_A_gather(cfw, COND, 1, 0, kc * 32, COND, lane);
          af1 = load_A_gather(cfw, COND, 1, 16, kc * 32, COND, lane);
          ag0 = load_A_gather(cgw, COND, 1, 0, kc * 32, COND, lane);
          ag1 = load_A_gather(cgw, COND, 1, 16, kc * 32, COND, lane);
        }
        bbuf = s_h; bk0 = kc * 32; bld = LDK_H;
      }
#pragma unroll
      for (int j = 0; j < MAXB; ++j) {
        if (blkid[j] >= NBLK) continue;
        const int cb = blkid[j] * 16;
        const v16bf bb = load_B(bbuf, bld, bk0, cb, lane);
        if (ks == 0) {
          accF[j][0] = wmma_bf(af0, bb, vzero);
          accF[j][1] = wmma_bf(af1, bb, vzero);
          accG[j][0] = wmma_bf(ag0, bb, vzero);
          accG[j][1] = wmma_bf(ag1, bb, vzero);
        } else {
          accF[j][0] = wmma_bf(af0, bb, accF[j][0]);
          accF[j][1] = wmma_bf(af1, bb, accF[j][1]);
          accG[j][0] = wmma_bf(ag0, bb, accG[j][0]);
          accG[j][1] = wmma_bf(ag1, bb, accG[j][1]);
        }
      }
    }
    __syncthreads();   // all reads of y from s_act complete

    // z = tanh(F) * sigmoid(G)  -> s_act
#pragma unroll
    for (int j = 0; j < MAXB; ++j) {
      if (blkid[j] >= NBLK) continue;
      const int cb = blkid[j] * 16;
#pragma unroll
      for (int mt = 0; mt < 2; ++mt) {
        v8f zz;
#pragma unroll
        for (int e = 0; e < 8; ++e)
          zz[e] = fast_tanh(accF[j][mt][e]) * fast_sigmoid(accG[j][mt][e]);
        store_C(s_act, LDK_A, mt * 16, cb, zz, lane);
      }
    }
    __syncthreads();

    // Phase 3: o = out_w @ z ; res += o ; skip += o ; mirror res to LDS
    {
      v16bf ao0, ao1;
      if constexpr (PACKED) {
        ao0 = load_A_packed(wp, tb + 20, lane);
        ao1 = load_A_packed(wp, tb + 21, lane);
      } else {
        ao0 = load_A_gather(ow, DC, 1, 0, 0, 32, lane);
        ao1 = load_A_gather(ow, DC, 1, 16, 0, 32, lane);
      }
#pragma unroll
      for (int j = 0; j < MAXB; ++j) {
        if (blkid[j] >= NBLK) continue;
        const int cb = blkid[j] * 16;
        const v16bf bz = load_B(s_act, LDK_A, 0, cb, lane);
        const v8f o0 = wmma_bf(ao0, bz, vzero);
        const v8f o1 = wmma_bf(ao1, bz, vzero);
        resC[j][0] += o0;  skipC[j][0] += o0;
        resC[j][1] += o1;  skipC[j][1] += o1;
        store_C(s_res, LDK_R, 0, cb + 1, resC[j][0], lane);
        store_C(s_res, LDK_R, 16, cb + 1, resC[j][1], lane);
      }
    }
    __syncthreads();   // res mirror visible before next layer's conv
  }

  // ---- end stack: logit = end2 @ relu(end1 @ relu(skips)) ----
  // s_h is dead; carve per-wave column-major [16][LDK_E] scratch from it.
  __bf16* ew = s_h + (size_t)wave * (16 * LDK_E);   // 4*8448B = 33792 <= 36608
  const int sc0 = wave * 16;   // per-wave column slice of s_act

#pragma unroll
  for (int j = 0; j < MAXB; ++j) {
    const int bb = blkid[j];
    if (bb < VBLK0 || bb >= NBLK) continue;   // valid (non-halo) blocks only
    const int cb = bb * 16;
    const int tg = tbase + cb;                // in [0, T)

    // relu(skips) -> per-wave slice (same-wave LDS ordering suffices)
#pragma unroll
    for (int mt = 0; mt < 2; ++mt) {
      v8f r;
#pragma unroll
      for (int e = 0; e < 8; ++e) r[e] = fmaxf(skipC[j][mt][e], 0.f);
      store_C(s_act, LDK_A, mt * 16, sc0, r, lane);
    }
    const v16bf bsk = load_B(s_act, LDK_A, 0, sc0, lane);

    // e = relu(end1_w @ rskip)  (256 x 16) -> ew
#pragma unroll 1
    for (int mt = 0; mt < 16; ++mt) {
      v16bf a;
      if constexpr (PACKED) a = load_A_packed(wp, TB_END1 + mt, lane);
      else                  a = load_A_gather(end1_w, RC, 1, mt * 16, 0, 32, lane);
      const v8f ee = wmma_bf(a, bsk, vzero);
      v8f er;
#pragma unroll
      for (int e = 0; e < 8; ++e) er[e] = fmaxf(ee[e], 0.f);
      store_C(ew, LDK_E, mt * 16, 0, er, lane);
    }

    // hoist the 8 B operands (K = 256) for this block
    v16bf b2[8];
#pragma unroll
    for (int ks = 0; ks < 8; ++ks) b2[ks] = load_B(ew, LDK_E, ks * 32, 0, lane);

    // logits = end2_w @ e  (256 x 16), store f32 to global
#pragma unroll 1
    for (int mt = 0; mt < 16; ++mt) {
      v8f acc = vzero;
#pragma unroll
      for (int ks = 0; ks < 8; ++ks) {
        v16bf a;
        if constexpr (PACKED) a = load_A_packed(wp, TB_END2 + mt * 8 + ks, lane);
        else                  a = load_A_gather(end2_w, SC, 1, mt * 16, ks * 32, 256, lane);
        acc = (ks == 0) ? wmma_bf(a, b2[ks], vzero) : wmma_bf(a, b2[ks], acc);
      }
#pragma unroll
      for (int e = 0; e < 8; ++e) {
        const int m = mt * 16 + g * 8 + e;
        out[((size_t)(bidx * CLS + m)) * T_ + tg + n] = acc[e];
      }
    }
  }
}

extern "C" void kernel_launch(void* const* d_in, const int* in_sizes, int n_in,
                              void* d_out, int out_size, void* d_ws, size_t ws_size,
                              hipStream_t stream) {
  (void)in_sizes; (void)n_in; (void)out_size;
  const float* x       = (const float*)d_in[0];
  const float* h       = (const float*)d_in[1];
  const float* start_w = (const float*)d_in[2];
  const float* dil_w   = (const float*)d_in[3];
  const float* f_w     = (const float*)d_in[4];
  const float* g_w     = (const float*)d_in[5];
  const float* cf_w    = (const float*)d_in[6];
  const float* cg_w    = (const float*)d_in[7];
  const float* out_w   = (const float*)d_in[8];
  const float* end1_w  = (const float*)d_in[9];
  const float* end2_w  = (const float*)d_in[10];
  float* out = (float*)d_out;

  dim3 grid(BATCH * NT);   // 512 workgroups
  dim3 block(NTHR);        // 4 wave32s

  if (ws_size >= WS_NEEDED && d_ws != nullptr) {
    __bf16* wp = (__bf16*)d_ws;
    hipLaunchKernelGGL(pack_weights_kernel, dim3(NTILES), dim3(32), 0, stream,
                       dil_w, f_w, g_w, cf_w, cg_w, out_w, end1_w, end2_w, wp);
    hipLaunchKernelGGL((wavenet_wmma_kernel<true>), grid, block, 0, stream,
                       x, h, start_w, dil_w, f_w, g_w, cf_w, cg_w, out_w,
                       end1_w, end2_w, wp, out);
  } else {
    hipLaunchKernelGGL((wavenet_wmma_kernel<false>), grid, block, 0, stream,
                       x, h, start_w, dil_w, f_w, g_w, cf_w, cg_w, out_w,
                       end1_w, end2_w, (const __bf16*)nullptr, out);
  }
}
